// MultiHeadAttentionQuantum_65481071398822
// MI455X (gfx1250) — matrix operand order repaired and checked by reading, not compiled, not run on hardware
//
#include <hip/hip_runtime.h>
#include <hip/hip_bf16.h>

// ---------------------------------------------------------------------------
// MI455X / gfx1250 fused MHA + quantum-head pipeline (wave32, bf16 WMMA)
// Shapes fixed by the reference: B=4, S=1024, E=128, H=16, d_k=8, 8 wires.
// ---------------------------------------------------------------------------

typedef __attribute__((ext_vector_type(16))) __bf16 v16bf;
typedef __attribute__((ext_vector_type(8)))  __bf16 v8bf;
typedef __attribute__((ext_vector_type(8)))  float  v8f;

union BF16x16 { v16bf v; v8bf h[2]; __bf16 e[16]; };

#define B_DIM 4
#define S_DIM 1024
#define E_DIM 128
#define H_DIM 16
#define DK    8
#define NTOK  (B_DIM * S_DIM)   // 4096

// workspace layout (bytes); all offsets 4KB-multiples. Total ~5.4 MB.
#define OFF_XBF  ((size_t)0)
#define OFF_WQ   ((size_t)NTOK * E_DIM * 2)            // 1,048,576
#define OFF_WK   (OFF_WQ + (size_t)E_DIM * E_DIM * 2)
#define OFF_WV   (OFF_WK + (size_t)E_DIM * E_DIM * 2)
#define OFF_WC   (OFF_WV + (size_t)E_DIM * E_DIM * 2)
#define OFF_Q    (OFF_WC + (size_t)E_DIM * E_DIM * 2)
#define OFF_K    (OFF_Q  + (size_t)NTOK * E_DIM * 2)
#define OFF_V    (OFF_K  + (size_t)NTOK * E_DIM * 2)
#define OFF_QO   (OFF_V  + (size_t)NTOK * E_DIM * 2)

// ---------------------------------------------------------------------------
// fp32 -> bf16 conversion
// ---------------------------------------------------------------------------
__global__ void cvt_f32_to_bf16_kernel(const float* __restrict__ src,
                                       __bf16* __restrict__ dst, int n) {
  int i = blockIdx.x * blockDim.x + threadIdx.x;
  if (i < n) dst[i] = (__bf16)src[i];
}

// ---------------------------------------------------------------------------
// One 16x16 WMMA output tile of  C = A(MxK, row-major bf16) @ W^T(N x K bf16)
// A-operand layout: lane(0..15)=row, halves split K (8-elem runs 0..7,16..23).
// B-operand layout: lane holds column n, halves split K in 8-elem runs, same order as A.
// ---------------------------------------------------------------------------
__device__ __forceinline__ v8f wmma_gemm_tile(const __bf16* __restrict__ A,
                                              const __bf16* __restrict__ W,
                                              int m0, int n0, int K,
                                              int lda, int ldw) {
  const int lane = threadIdx.x & 31;
  const int li = lane & 15;
  const int hh = lane >> 4;
  v8f acc = {};
#pragma unroll
  for (int k0 = 0; k0 < K; k0 += 32) {
    BF16x16 a, b;
    const __bf16* ap = A + (size_t)(m0 + li) * lda + k0 + 8 * hh;
    a.h[0] = *(const v8bf*)(ap);        // K = k0+8h .. +7
    a.h[1] = *(const v8bf*)(ap + 16);   // K = k0+8h+16 .. +23
    const __bf16* bp = W + (size_t)(n0 + li) * ldw + k0 + 8 * hh;
    b.h[0] = *(const v8bf*)(bp);        // K = k0+8h .. +7
    b.h[1] = *(const v8bf*)(bp + 16);   // K = k0+8h+16 .. +23
    acc = __builtin_amdgcn_wmma_f32_16x16x32_bf16(false, a.v, false, b.v,
                                                  (short)0, acc, false, false);
  }
  return acc;
}

// ---------------------------------------------------------------------------
// Projection: out[b,h,s,d] = sum_e x[t,e] * W[h*8+d, e]   (bf16 out, head-major)
// grid (NTOK/32, E/64), block 256 (8 waves, 2x4 tile grid of 16x16 tiles).
// ---------------------------------------------------------------------------
__global__ __launch_bounds__(256) void proj_qkv_kernel(
    const __bf16* __restrict__ xbf, const __bf16* __restrict__ wbf,
    __bf16* __restrict__ out) {
  const int wave = threadIdx.x >> 5;
  const int m0 = blockIdx.x * 32 + (wave & 1) * 16;
  const int n0 = blockIdx.y * 64 + (wave >> 1) * 16;
  v8f acc = wmma_gemm_tile(xbf, wbf, m0, n0, E_DIM, E_DIM, E_DIM);
  const int lane = threadIdx.x & 31;
  const int li = lane & 15, hh = lane >> 4;
  const int j = n0 + li;
  const int h = j >> 3, d = j & 7;
#pragma unroll
  for (int r = 0; r < 8; ++r) {
    const int t = m0 + r + 8 * hh;
    const int b = t >> 10, s = t & (S_DIM - 1);
    out[((((size_t)b * H_DIM + h) * S_DIM + s) * DK) + d] = (__bf16)acc[r];
  }
}

// ---------------------------------------------------------------------------
// Final projection: d_out[t, j] = sum_e qout[t,e] * Wc[j,e]   (fp32 out)
// ---------------------------------------------------------------------------
__global__ __launch_bounds__(256) void out_proj_kernel(
    const __bf16* __restrict__ qob, const __bf16* __restrict__ wcb,
    float* __restrict__ dst) {
  const int wave = threadIdx.x >> 5;
  const int m0 = blockIdx.x * 32 + (wave & 1) * 16;
  const int n0 = blockIdx.y * 64 + (wave >> 1) * 16;
  v8f acc = wmma_gemm_tile(qob, wcb, m0, n0, E_DIM, E_DIM, E_DIM);
  const int lane = threadIdx.x & 31;
  const int li = lane & 15, hh = lane >> 4;
#pragma unroll
  for (int r = 0; r < 8; ++r) {
    const int t = m0 + r + 8 * hh;
    dst[(size_t)t * E_DIM + n0 + li] = acc[r];
  }
}

// ---------------------------------------------------------------------------
// Fused flash-attention + quantum readout.
// grid (S/128, B*H), block 256 = 8 waves; each wave owns 16 query rows.
// Scores: WMMA with d_k=8 zero-padded into the K=32 slot. Online softmax in
// fp32 VALU with width-16 shuffles. P transposed C->A layout via per-wave LDS.
// V rows staged through LDS: one global_load_b128 per lane per 32-row chunk,
// then the strided B-operand gather hits the 64-bank LDS instead of HBM/L2.
// ---------------------------------------------------------------------------
__global__ __launch_bounds__(256) void attn_quantum_kernel(
    const __bf16* __restrict__ Q, const __bf16* __restrict__ K,
    const __bf16* __restrict__ V, const float* __restrict__ params,
    __bf16* __restrict__ qout) {
  __shared__ alignas(16) __bf16 Plds[8][16 * 32];   // per-wave 16x32 P tile
  __shared__ alignas(16) __bf16 Vlds[8][32][DK];    // per-wave 32-row V chunk
  __shared__ float Wlds[8][16][DK];                 // per-wave weighted rows

  const int wave = threadIdx.x >> 5;
  const int lane = threadIdx.x & 31;
  const int li = lane & 15;
  const int hh = lane >> 4;
  const int bh = blockIdx.y;
  const int b = bh >> 4, h = bh & 15;
  const int qbase = blockIdx.x * 128 + wave * 16;   // query row within S

  const __bf16* Qp = Q + (size_t)bh * S_DIM * DK;
  const __bf16* Kp = K + (size_t)bh * S_DIM * DK;
  const __bf16* Vp = V + (size_t)bh * S_DIM * DK;

  // Q A-operand: rows 0..15 in lanes, K-dim (d_k=8) padded to 32 with zeros.
  BF16x16 qa;
#pragma unroll
  for (int i = 0; i < 16; ++i) qa.e[i] = (__bf16)0.f;
  if (hh == 0) qa.h[0] = *(const v8bf*)(Qp + (size_t)(qbase + li) * DK);

  float mrow[8], lrow[8];
  v8f O = {};
#pragma unroll
  for (int r = 0; r < 8; ++r) { mrow[r] = -1e30f; lrow[r] = 0.f; }

  const float scale = 0.35355339059327373f;  // 1/sqrt(d_k)

  for (int kv = 0; kv < S_DIM; kv += 32) {
    if (kv + 32 < S_DIM) {  // hint next K/V chunk toward the caches
      __builtin_prefetch(Kp + (size_t)(kv + 32 + lane) * DK, 0, 1);
      __builtin_prefetch(Vp + (size_t)(kv + 32 + lane) * DK, 0, 1);
    }
    // K^T B-operands for columns [kv,kv+16) and [kv+16,kv+32)
    BF16x16 kb0, kb1;
#pragma unroll
    for (int i = 0; i < 16; ++i) { kb0.e[i] = (__bf16)0.f; kb1.e[i] = (__bf16)0.f; }
    if (hh == 0) {
      kb0.h[0] = *(const v8bf*)(Kp + (size_t)(kv + li) * DK);
      kb1.h[0] = *(const v8bf*)(Kp + (size_t)(kv + 16 + li) * DK);
    }
    // V chunk: one 16B row per lane, parked in LDS during the transpose window
    const v8bf vrow = *(const v8bf*)(Vp + (size_t)(kv + lane) * DK);

    v8f s0 = {}, s1 = {};
    s0 = __builtin_amdgcn_wmma_f32_16x16x32_bf16(false, qa.v, false, kb0.v,
                                                 (short)0, s0, false, false);
    s1 = __builtin_amdgcn_wmma_f32_16x16x32_bf16(false, qa.v, false, kb1.v,
                                                 (short)0, s1, false, false);
#pragma unroll
    for (int r = 0; r < 8; ++r) { s0[r] *= scale; s1[r] *= scale; }

    // online softmax: row = r + 8*hh, its 32 columns live across a 16-lane half
    float alpha[8];
#pragma unroll
    for (int r = 0; r < 8; ++r) {
      float v = fmaxf(s0[r], s1[r]);
      for (int off = 1; off < 16; off <<= 1) v = fmaxf(v, __shfl_xor(v, off, 16));
      const float mnew = fmaxf(mrow[r], v);
      alpha[r] = __expf(mrow[r] - mnew);
      mrow[r] = mnew;
    }
    v8f p0, p1;
#pragma unroll
    for (int r = 0; r < 8; ++r) {
      p0[r] = __expf(s0[r] - mrow[r]);
      p1[r] = __expf(s1[r] - mrow[r]);
      float sum = p0[r] + p1[r];
      for (int off = 1; off < 16; off <<= 1) sum += __shfl_xor(sum, off, 16);
      lrow[r] = lrow[r] * alpha[r] + sum;
      O[r] *= alpha[r];
    }

    // transpose P (C layout: lane=col, elem=row) -> A layout via per-wave LDS;
    // stage the V chunk in the same sync window.
    __syncthreads();
    __bf16* pl = Plds[wave];
#pragma unroll
    for (int r = 0; r < 8; ++r) {
      const int row = r + 8 * hh;
      pl[row * 32 + li]      = (__bf16)p0[r];
      pl[row * 32 + 16 + li] = (__bf16)p1[r];
    }
    *(v8bf*)(&Vlds[wave][lane][0]) = vrow;
    __syncthreads();

    BF16x16 pa;
    {
      const __bf16* pp = pl + li * 32 + 8 * hh;
      pa.h[0] = *(const v8bf*)(pp);        // cols 8h .. 8h+7
      pa.h[1] = *(const v8bf*)(pp + 16);   // cols 8h+16 .. 8h+23
    }
    // V B-operand from LDS: K-dim = 32 kv rows, N-dim = d (cols >=8 zero)
    BF16x16 vb;
#pragma unroll
    for (int e = 0; e < 16; ++e) {
      vb.e[e] = (li < DK) ? Vlds[wave][(e < 8) ? (8 * hh + e) : (16 + 8 * hh + (e - 8))][li] : (__bf16)0.f;
    }
    O = __builtin_amdgcn_wmma_f32_16x16x32_bf16(false, pa.v, false, vb.v,
                                                (short)0, O, false, false);
  }

  // normalize by softmax denominator
#pragma unroll
  for (int r = 0; r < 8; ++r) {
    const float inv = (lrow[r] > 0.f) ? (1.0f / lrow[r]) : 0.f;
    O[r] *= inv;
  }

  // transpose weighted rows to per-token layout for the quantum readout
  __syncthreads();
  if (li < DK) {
#pragma unroll
    for (int r = 0; r < 8; ++r) Wlds[wave][r + 8 * hh][li] = O[r];
  }
  __syncthreads();

  // quantum heads: product state => <Z_w> = prefix products of per-wire z.
  // CNOT-ring conjugation: Z_w -> Z_0..Z_w (w>=1), Z_0 -> Z_1..Z_7.
  if (lane < 16) {
    const int srow = qbase + lane;
    const float* pp = params + h * 24;  // [8 wires][3]
    float z[8];
#pragma unroll
    for (int w = 0; w < 8; ++w) {
      const float x   = Wlds[wave][lane][w];
      const float hx  = 0.5f * x;
      const float p0a = 0.5f * pp[w * 3 + 0];
      const float p1a = 0.5f * pp[w * 3 + 1];
      const float p2a = 0.5f * pp[w * 3 + 2];
      const float a0r = __cosf(hx), a1i = -__sinf(hx);      // RX(x)|0>
      const float c  = __cosf(p0a), s  = __sinf(p0a);       // RY(p0)
      const float b0r = c * a0r, b0i = -s * a1i;
      const float b1r = s * a0r, b1i =  c * a1i;
      const float cp = __cosf(p1a), sp = __sinf(p1a);       // RZ(p1)
      const float c0r = cp * b0r + sp * b0i, c0i = cp * b0i - sp * b0r;
      const float c1r = cp * b1r - sp * b1i, c1i = cp * b1i + sp * b1r;
      const float c2 = __cosf(p2a), s2 = __sinf(p2a);       // RX(p2)
      const float v0r = c2 * c0r + s2 * c1i, v0i = c2 * c0i - s2 * c1r;
      const float v1r = s2 * c0i + c2 * c1r, v1i = -s2 * c0r + c2 * c1i;
      z[w] = (v0r * v0r + v0i * v0i) - (v1r * v1r + v1i * v1i);
    }
    float outv[8];
    float tail = 1.f;
#pragma unroll
    for (int u = 1; u < 8; ++u) tail *= z[u];
    outv[0] = tail;                       // <Z_0> = z1*...*z7
    float pref = z[0];
#pragma unroll
    for (int w = 1; w < 8; ++w) { pref *= z[w]; outv[w] = pref; }

    __bf16* qo = qout + (((size_t)(b * S_DIM + srow)) * H_DIM + h) * DK;
#pragma unroll
    for (int w = 0; w < 8; ++w) qo[w] = (__bf16)outv[w];
  }
}

// ---------------------------------------------------------------------------
// launcher
// ---------------------------------------------------------------------------
extern "C" void kernel_launch(void* const* d_in, const int* in_sizes, int n_in,
                              void* d_out, int out_size, void* d_ws, size_t ws_size,
                              hipStream_t stream) {
  const float* x      = (const float*)d_in[0];
  const float* Wq     = (const float*)d_in[1];
  const float* Wk     = (const float*)d_in[2];
  const float* Wv     = (const float*)d_in[3];
  const float* Wc     = (const float*)d_in[4];
  const float* params = (const float*)d_in[5];

  char* ws = (char*)d_ws;
  __bf16* xbf = (__bf16*)(ws + OFF_XBF);
  __bf16* wqb = (__bf16*)(ws + OFF_WQ);
  __bf16* wkb = (__bf16*)(ws + OFF_WK);
  __bf16* wvb = (__bf16*)(ws + OFF_WV);
  __bf16* wcb = (__bf16*)(ws + OFF_WC);
  __bf16* Qb  = (__bf16*)(ws + OFF_Q);
  __bf16* Kb  = (__bf16*)(ws + OFF_K);
  __bf16* Vb  = (__bf16*)(ws + OFF_V);
  __bf16* QOb = (__bf16*)(ws + OFF_QO);

  const int nx = NTOK * E_DIM;      // 524288
  const int nw = E_DIM * E_DIM;     // 16384
  cvt_f32_to_bf16_kernel<<<(nx + 255) / 256, 256, 0, stream>>>(x,  xbf, nx);
  cvt_f32_to_bf16_kernel<<<(nw + 255) / 256, 256, 0, stream>>>(Wq, wqb, nw);
  cvt_f32_to_bf16_kernel<<<(nw + 255) / 256, 256, 0, stream>>>(Wk, wkb, nw);
  cvt_f32_to_bf16_kernel<<<(nw + 255) / 256, 256, 0, stream>>>(Wv, wvb, nw);
  cvt_f32_to_bf16_kernel<<<(nw + 255) / 256, 256, 0, stream>>>(Wc, wcb, nw);

  dim3 pg(NTOK / 32, E_DIM / 64);   // (128, 2)
  proj_qkv_kernel<<<pg, 256, 0, stream>>>(xbf, wqb, Qb);
  proj_qkv_kernel<<<pg, 256, 0, stream>>>(xbf, wkb, Kb);
  proj_qkv_kernel<<<pg, 256, 0, stream>>>(xbf, wvb, Vb);

  attn_quantum_kernel<<<dim3(S_DIM / 128, B_DIM * H_DIM), 256, 0, stream>>>(
      Qb, Kb, Vb, params, QOb);

  out_proj_kernel<<<pg, 256, 0, stream>>>(QOb, wcb, (float*)d_out);
}
